// SimpleMoE_1520418423055
// MI455X (gfx1250) — matrix operand order repaired and checked by reading, not compiled, not run on hardware
//
#include <hip/hip_runtime.h>
#include <hip/hip_bf16.h>

#define D_DIM 1024
#define F_DIM 2816
#define E_NUM 8
#define T_NUM 16384
#define TM 32            // token tile height per block
#define XS_STRIDE 1032   // 1024 + 8 (bank-conflict padding, keeps 16B alignment)
#define HS_STRIDE 264    // 256 + 8
#define FS_CHUNK 256     // F superchunk (8 waves x 32)
#define WELEMS ((size_t)E_NUM * F_DIM * D_DIM)   // elems per weight tensor

typedef __attribute__((ext_vector_type(16))) __bf16 v16bf;
typedef __attribute__((ext_vector_type(8)))  __bf16 v8bf;
typedef __attribute__((ext_vector_type(8)))  float  v8f;

// WMMA A-operand (16x32 bf16) from row-major bf16 LDS tile.
// lanes 0-15: row=lane, K = k0+[0..15]
// lanes16-31: row=lane-16, K = k0+[16..31]
__device__ inline v16bf lds_load_a(const __bf16* base, int stride, int lane, int k0) {
  const int row  = lane & 15;
  const int koff = k0 + ((lane & 16) ? 16 : 0);
  const __bf16* p = base + row * stride + koff;
  v8bf lo = *(const v8bf*)(p);
  v8bf hi = *(const v8bf*)(p + 8);
  return __builtin_shufflevector(lo, hi, 0,1,2,3,4,5,6,7,8,9,10,11,12,13,14,15);
}

// WMMA B-operand from pre-converted bf16 weights: 16 contiguous K values.
__device__ inline v16bf gbl_load_b_bf16(const __bf16* __restrict__ p) {
  v8bf lo = *(const v8bf*)(p);
  v8bf hi = *(const v8bf*)(p + 8);
  return __builtin_shufflevector(lo, hi, 0,1,2,3,4,5,6,7,8,9,10,11,12,13,14,15);
}

// WMMA B-operand from fp32 weights (fallback when ws too small): convert on the fly.
__device__ inline v16bf gbl_load_b_f32(const float* __restrict__ p) {
  const float4* q = (const float4*)p;
  float4 f0 = q[0], f1 = q[1], f2 = q[2], f3 = q[3];
  v16bf b;
  b[0]=(__bf16)f0.x; b[1]=(__bf16)f0.y; b[2]=(__bf16)f0.z; b[3]=(__bf16)f0.w;
  b[4]=(__bf16)f1.x; b[5]=(__bf16)f1.y; b[6]=(__bf16)f1.z; b[7]=(__bf16)f1.w;
  b[8]=(__bf16)f2.x; b[9]=(__bf16)f2.y; b[10]=(__bf16)f2.z; b[11]=(__bf16)f2.w;
  b[12]=(__bf16)f3.x; b[13]=(__bf16)f3.y; b[14]=(__bf16)f3.z; b[15]=(__bf16)f3.w;
  return b;
}

// ---------------- Weight pre-conversion: fp32 -> bf16, 8 elems/thread ----------------
__global__ __launch_bounds__(256) void cvt_bf16_kernel(const float* __restrict__ src,
                                                       __bf16* __restrict__ dst, int n8) {
  int i = blockIdx.x * 256 + threadIdx.x;
  if (i >= n8) return;
  const float4* s = (const float4*)src + (size_t)i * 2;
  float4 a = s[0], b = s[1];
  v8bf o;
  o[0]=(__bf16)a.x; o[1]=(__bf16)a.y; o[2]=(__bf16)a.z; o[3]=(__bf16)a.w;
  o[4]=(__bf16)b.x; o[5]=(__bf16)b.y; o[6]=(__bf16)b.z; o[7]=(__bf16)b.w;
  ((v8bf*)dst)[i] = o;
}

// ---------------- Router: one wave32 per token ----------------
__global__ __launch_bounds__(256) void moe_router_kernel(
    const float* __restrict__ x, const float* __restrict__ gw,
    int* __restrict__ cnt, int* __restrict__ btok, float* __restrict__ bwt)
{
  const int lane = threadIdx.x & 31;
  const int t = blockIdx.x * 8 + (threadIdx.x >> 5);
  float acc[E_NUM];
#pragma unroll
  for (int e = 0; e < E_NUM; ++e) acc[e] = 0.0f;
  const float* xr = x + (size_t)t * D_DIM;
  for (int d = lane; d < D_DIM; d += 32) {
    float xv = xr[d];
#pragma unroll
    for (int e = 0; e < E_NUM; ++e) acc[e] += xv * gw[e * D_DIM + d];
  }
#pragma unroll
  for (int e = 0; e < E_NUM; ++e) {
#pragma unroll
    for (int off = 16; off > 0; off >>= 1)
      acc[e] += __shfl_down(acc[e], off, 32);
  }
  if (lane == 0) {
    float v0 = -3.0e38f, v1 = -3.0e38f; int i0 = 0, i1 = 0;
#pragma unroll
    for (int e = 0; e < E_NUM; ++e) {
      float v = acc[e];
      if (v > v0) { v1 = v0; i1 = i0; v0 = v; i0 = e; }
      else if (v > v1) { v1 = v; i1 = e; }
    }
    float e1 = __expf(v1 - v0);
    float inv = 1.0f / (1.0f + e1);   // cold path: exact divide is fine
    int p0 = atomicAdd(&cnt[i0], 1);
    btok[i0 * T_NUM + p0] = t; bwt[i0 * T_NUM + p0] = inv;
    int p1 = atomicAdd(&cnt[i1], 1);
    btok[i1 * T_NUM + p1] = t; bwt[i1 * T_NUM + p1] = e1 * inv;
  }
}

// ---------------- FFN: one block = 32 tokens of expert e ----------------
// BF16W: weights are pre-converted bf16 in workspace; else fp32 converted on the fly.
template <bool BF16W>
__global__ __launch_bounds__(256) void moe_ffn_kernel(
    const float* __restrict__ x,  const void* __restrict__ w1v,
    const void* __restrict__ w3v, const void* __restrict__ w2v,
    const int* __restrict__ cnt,  const int* __restrict__ btok,
    const float* __restrict__ bwt, float* __restrict__ out, int e)
{
  __shared__ __attribute__((aligned(16))) __bf16 Xs[TM * XS_STRIDE];  // 66 KB
  __shared__ __attribute__((aligned(16))) __bf16 Hs[TM * HS_STRIDE];  // 16.9 KB
  __shared__ int    tokS[TM];
  __shared__ float  wtS[TM];

  const int count = cnt[e];
  const int tileBase = blockIdx.x * TM;
  if (tileBase >= count) return;

  const int tid  = threadIdx.x;
  const int lane = tid & 31;
  const int wv   = tid >> 5;               // wave 0..7
  const int bn   = lane & 15;              // B/C column within 16-tile
  const int bko  = (lane & 16) ? 16 : 0;   // B K half select
  const int mb   = (lane & 16) ? 8 : 0;    // C/D row base

  if (tid < TM) {
    int idx = tileBase + tid;
    if (idx < count) { tokS[tid] = btok[e * T_NUM + idx]; wtS[tid] = bwt[e * T_NUM + idx]; }
    else             { tokS[tid] = 0;                      wtS[tid] = 0.0f; }
  }
  __syncthreads();

  // X tile [32,1024] fp32 -> bf16 LDS
  for (int i = tid; i < TM * 256; i += 256) {
    const int row = i >> 8, c4 = i & 255;
    const float4 v = ((const float4*)(x + (size_t)tokS[row] * D_DIM))[c4];
    __bf16* dst = &Xs[row * XS_STRIDE + c4 * 4];
    dst[0] = (__bf16)v.x; dst[1] = (__bf16)v.y; dst[2] = (__bf16)v.z; dst[3] = (__bf16)v.w;
  }
  __syncthreads();

  v8f zero = {};
  v8f acc[2][8];                           // O[rowhalf][ntile]: 32 x (wv*128..+127)
#pragma unroll
  for (int h = 0; h < 2; ++h)
#pragma unroll
    for (int i = 0; i < 8; ++i) acc[h][i] = zero;

  const size_t wb13 = (size_t)e * F_DIM * D_DIM;   // w1/w3 [E,F,D]
  const size_t wb2  = (size_t)e * D_DIM * F_DIM;   // w2    [E,D,F]

  for (int fs = 0; fs < F_DIM; fs += FS_CHUNK) {
    // Phase A: this wave computes H[0:32, wv*32 .. wv*32+31]
    const int f0 = fs + wv * 32;
    v8f a1[2][2], a3[2][2];                // [rowhalf][ftile]
#pragma unroll
    for (int h = 0; h < 2; ++h)
#pragma unroll
      for (int t = 0; t < 2; ++t) { a1[h][t] = zero; a3[h][t] = zero; }

    for (int k0 = 0; k0 < D_DIM; k0 += 32) {
      v16bf av0 = lds_load_a(Xs, XS_STRIDE, lane, k0);
      v16bf av1 = lds_load_a(Xs + 16 * XS_STRIDE, XS_STRIDE, lane, k0);
#pragma unroll
      for (int t = 0; t < 2; ++t) {
        const size_t frow = wb13 + (size_t)(f0 + t * 16 + bn) * D_DIM + (k0 + bko);
        v16bf b1, b3;
        if constexpr (BF16W) {
          b1 = gbl_load_b_bf16((const __bf16*)w1v + frow);
          b3 = gbl_load_b_bf16((const __bf16*)w3v + frow);
        } else {
          b1 = gbl_load_b_f32((const float*)w1v + frow);
          b3 = gbl_load_b_f32((const float*)w3v + frow);
        }
        // each B feeds both row halves (2x reuse)
        a1[0][t] = __builtin_amdgcn_wmma_f32_16x16x32_bf16(false, av0, false, b1, (short)0, a1[0][t], false, false);
        a1[1][t] = __builtin_amdgcn_wmma_f32_16x16x32_bf16(false, av1, false, b1, (short)0, a1[1][t], false, false);
        a3[0][t] = __builtin_amdgcn_wmma_f32_16x16x32_bf16(false, av0, false, b3, (short)0, a3[0][t], false, false);
        a3[1][t] = __builtin_amdgcn_wmma_f32_16x16x32_bf16(false, av1, false, b3, (short)0, a3[1][t], false, false);
      }
    }
    // h = silu(x@w1^T) * (x@w3^T) -> shared bf16 (C/D layout: M = r+mb, N = bn)
#pragma unroll
    for (int h = 0; h < 2; ++h)
#pragma unroll
      for (int t = 0; t < 2; ++t)
#pragma unroll
        for (int r = 0; r < 8; ++r) {
          float g = a1[h][t][r];
          float hv = g * __builtin_amdgcn_rcpf(1.0f + __expf(-g)) * a3[h][t][r];
          Hs[(h * 16 + r + mb) * HS_STRIDE + wv * 32 + t * 16 + bn] = (__bf16)hv;
        }
    __syncthreads();

    // Phase B: O += H[:, fs..fs+256] @ w2[:, fs..fs+256]^T (wave owns 128 d-cols)
    for (int kf = 0; kf < FS_CHUNK; kf += 32) {
      v16bf av0 = lds_load_a(Hs, HS_STRIDE, lane, kf);
      v16bf av1 = lds_load_a(Hs + 16 * HS_STRIDE, HS_STRIDE, lane, kf);
#pragma unroll
      for (int nt = 0; nt < 8; ++nt) {
        const int d = wv * 128 + nt * 16 + bn;
        const size_t off = wb2 + (size_t)d * F_DIM + (fs + kf + bko);
        v16bf b2;
        if constexpr (BF16W) b2 = gbl_load_b_bf16((const __bf16*)w2v + off);
        else                 b2 = gbl_load_b_f32((const float*)w2v + off);
        acc[0][nt] = __builtin_amdgcn_wmma_f32_16x16x32_bf16(false, av0, false, b2, (short)0, acc[0][nt], false, false);
        acc[1][nt] = __builtin_amdgcn_wmma_f32_16x16x32_bf16(false, av1, false, b2, (short)0, acc[1][nt], false, false);
      }
    }
    __syncthreads();
  }

  // out[token, d] += wt * O  (race-free: experts run in separate sequential kernels)
#pragma unroll
  for (int h = 0; h < 2; ++h)
#pragma unroll
    for (int nt = 0; nt < 8; ++nt) {
      const int d = wv * 128 + nt * 16 + bn;
#pragma unroll
      for (int r = 0; r < 8; ++r) {
        const int m = h * 16 + r + mb;
        if (tileBase + m < count) {
          float* o = out + (size_t)tokS[m] * D_DIM + d;
          *o += acc[h][nt][r] * wtS[m];
        }
      }
    }
}

extern "C" void kernel_launch(void* const* d_in, const int* in_sizes, int n_in,
                              void* d_out, int out_size, void* d_ws, size_t ws_size,
                              hipStream_t stream) {
  const float* x  = (const float*)d_in[0];
  const float* gw = (const float*)d_in[1];
  const float* w1 = (const float*)d_in[2];
  const float* w3 = (const float*)d_in[3];
  const float* w2 = (const float*)d_in[4];
  float* out = (float*)d_out;

  int*   cnt  = (int*)d_ws;
  int*   btok = cnt + 64;
  float* bwt  = (float*)(btok + E_NUM * T_NUM);

  // bf16 weight mirror lives after the (<2MB) bucket region
  const size_t bucketEnd = (size_t)2 << 20;
  const size_t wbytes = WELEMS * 2;
  const bool preconv = ws_size >= bucketEnd + 3 * wbytes;
  __bf16* w1b = (__bf16*)((char*)d_ws + bucketEnd);
  __bf16* w3b = w1b + WELEMS;
  __bf16* w2b = w3b + WELEMS;

  hipMemsetAsync(cnt, 0, 64 * sizeof(int), stream);
  hipMemsetAsync(out, 0, (size_t)out_size * sizeof(float), stream);

  if (preconv) {
    const int n8 = (int)(WELEMS / 8);
    const int g  = (n8 + 255) / 256;
    cvt_bf16_kernel<<<g, 256, 0, stream>>>(w1, w1b, n8);
    cvt_bf16_kernel<<<g, 256, 0, stream>>>(w3, w3b, n8);
    cvt_bf16_kernel<<<g, 256, 0, stream>>>(w2, w2b, n8);
  }

  moe_router_kernel<<<T_NUM / 8, 256, 0, stream>>>(x, gw, cnt, btok, bwt);

  for (int e = 0; e < E_NUM; ++e) {
    if (preconv)
      moe_ffn_kernel<true><<<T_NUM / TM, 256, 0, stream>>>(x, w1b, w3b, w2b, cnt, btok, bwt, out, e);
    else
      moe_ffn_kernel<false><<<T_NUM / TM, 256, 0, stream>>>(x, w1, w3, w2, cnt, btok, bwt, out, e);
  }
}
